// EGATlayer_56281251446785
// MI455X (gfx1250) — matrix operand order repaired and checked by reading, not compiled, not run on hardware
//
#include <hip/hip_runtime.h>
#include <hip/hip_bf16.h>

#define NN 4096
#define CC 256

typedef __attribute__((ext_vector_type(16))) _Float16 v16h;
typedef __attribute__((ext_vector_type(8)))  float    v8f;
typedef __attribute__((ext_vector_type(4)))  unsigned int u32x4;

__device__ __forceinline__ float wave_max(float v) {
  #pragma unroll
  for (int m = 16; m >= 1; m >>= 1) v = fmaxf(v, __shfl_xor(v, m, 32));
  return v;
}
__device__ __forceinline__ float wave_sum(float v) {
  #pragma unroll
  for (int m = 16; m >= 1; m >>= 1) v += __shfl_xor(v, m, 32);
  return v;
}

// assemble a v16h WMMA fragment from two 16-byte chunks
__device__ __forceinline__ v16h load_v16h(const _Float16* p0, const _Float16* p1) {
  union { v16h h; u32x4 u[2]; } t;
  t.u[0] = *reinterpret_cast<const u32x4*>(p0);
  t.u[1] = *reinterpret_cast<const u32x4*>(p1);
  return t.h;
}

// ---------------------------------------------------------------- kernel 1a
// qk[i] = (x_i.Wq)(x_i.Wk) / (C * sqrt(3));  one wave per row
__global__ __launch_bounds__(256) void k_qk(const float* __restrict__ X,
                                            const float* __restrict__ Wq,
                                            const float* __restrict__ Wk,
                                            float* __restrict__ qk) {
  int lane = threadIdx.x & 31, w = threadIdx.x >> 5;
  int row = blockIdx.x * 8 + w;
  const float* xr = X + (size_t)row * CC;
  float sq = 0.f, sk = 0.f;
  #pragma unroll
  for (int c = lane; c < CC; c += 32) {
    float x = xr[c];
    sq += x * Wq[c];
    sk += x * Wk[c];
  }
  sq = wave_sum(sq);
  sk = wave_sum(sk);
  if (lane == 0) qk[row] = sq * sk * (1.0f / (256.0f * 1.7320508075688772f));
}

// ---------------------------------------------------------------- kernel 1b
// vT[f][j] = (X[j,:].Wv[:,f]) / 16  (f16, transposed for WMMA B fragments)
__global__ __launch_bounds__(256) void k_v(const float* __restrict__ X,
                                           const float* __restrict__ Wv,
                                           _Float16* __restrict__ vT) {
  __shared__ float xs[16][CC];
  int t = threadIdx.x;
  int jb = blockIdx.x * 16;
  for (int e = t; e < 16 * CC; e += 256)
    xs[e >> 8][e & 255] = X[(size_t)jb * CC + e];
  __syncthreads();
  int f = t;
  float acc[16];
  #pragma unroll
  for (int r = 0; r < 16; ++r) acc[r] = 0.f;
  for (int c = 0; c < CC; ++c) {
    float wv = Wv[(size_t)c * CC + f];
    #pragma unroll
    for (int r = 0; r < 16; ++r) acc[r] += xs[r][c] * wv;
  }
  union { _Float16 h[16]; u32x4 u[2]; } pk;
  #pragma unroll
  for (int r = 0; r < 16; ++r) pk.h[r] = (_Float16)(acc[r] * 0.0625f);
  u32x4* dst = reinterpret_cast<u32x4*>(vT + (size_t)f * NN + jb);
  dst[0] = pk.u[0];
  dst[1] = pk.u[1];
}

// ---------------------------------------------------------------- kernel 2
// Fused scores -> online softmax -> att@v (WMMA) -> ELU update.
// Block = 16 rows, 4 waves; wave w owns output columns [64w, 64w+64).
__global__ __launch_bounds__(128) void k_attn(const float* __restrict__ X,
                                              const float* __restrict__ dist,
                                              const float* __restrict__ bond,
                                              const float* __restrict__ deg,
                                              const float* __restrict__ qk,
                                              const _Float16* __restrict__ vT,
                                              const float* __restrict__ wb_p,
                                              const float* __restrict__ wg_p,
                                              float* __restrict__ outU,
                                              _Float16* __restrict__ U16) {
  __shared__ float    sS[16 * 128];
  __shared__ _Float16 sP[16 * 136];     // padded stride: bank-conflict-free frags
  __shared__ float    sM[16], sL[16], sA[16], sQK[16];
  int t = threadIdx.x, lane = t & 31, w = t >> 5;
  int ibase = blockIdx.x * 16;
  float wb = *wb_p, wg = *wg_p;
  if (t < 16) { sQK[t] = qk[ibase + t]; sM[t] = -__builtin_inff(); sL[t] = 0.f; }
  v8f acc[4] = {};
  int m    = lane & 15;
  int aoff = (lane < 16) ? 0 : 8;    // A-fragment K chunk offset
  int koff = (lane < 16) ? 0 : 8;    // B-fragment K offset
  __syncthreads();

  for (int jt = 0; jt < NN; jt += 128) {
    // --- phase A: scores for 16x128 tile (thread = one column)
    int j = jt + t;
    #pragma unroll
    for (int i = 0; i < 16; ++i) {
      size_t idx = (size_t)(ibase + i) * NN + j;
      float raw = sQK[i] + wb * bond[idx] + wg * dist[idx];
      float s = raw > 0.f ? raw : 0.2f * raw;            // leaky_relu(0.2)
      s += (deg[idx] > 0.f) ? 0.f : -1e9f;               // connectivity mask
      sS[i * 128 + t] = s;
    }
    __syncthreads();

    // --- phase B: online softmax; wave w handles rows 4w..4w+3
    #pragma unroll
    for (int rr = 0; rr < 4; ++rr) {
      int i = w * 4 + rr;
      float s0 = sS[i * 128 + lane];
      float s1 = sS[i * 128 + lane + 32];
      float s2 = sS[i * 128 + lane + 64];
      float s3 = sS[i * 128 + lane + 96];
      float mx = wave_max(fmaxf(fmaxf(s0, s1), fmaxf(s2, s3)));
      float m_old = sM[i];
      float m_new = fmaxf(m_old, mx);
      float p0 = __expf(s0 - m_new), p1 = __expf(s1 - m_new);
      float p2 = __expf(s2 - m_new), p3 = __expf(s3 - m_new);
      sP[i * 136 + lane]      = (_Float16)p0;
      sP[i * 136 + lane + 32] = (_Float16)p1;
      sP[i * 136 + lane + 64] = (_Float16)p2;
      sP[i * 136 + lane + 96] = (_Float16)p3;
      float ts = wave_sum(p0 + p1 + p2 + p3);
      if (lane == 0) {
        float alpha = __expf(m_old - m_new);
        sA[i] = alpha;
        sL[i] = sL[i] * alpha + ts;
        sM[i] = m_new;
      }
    }
    __syncthreads();

    // --- phase C: rescale accumulators, then 16 WMMAs over the K=128 tile
    float al[8];
    #pragma unroll
    for (int r = 0; r < 8; ++r) al[r] = sA[(lane < 16) ? r : r + 8];
    #pragma unroll
    for (int nt = 0; nt < 4; ++nt)
      #pragma unroll
      for (int r = 0; r < 8; ++r) acc[nt][r] *= al[r];

    #pragma unroll
    for (int kk = 0; kk < 4; ++kk) {
      const _Float16* ap = &sP[m * 136 + kk * 32 + aoff];
      v16h a = load_v16h(ap, ap + 16);
      #pragma unroll
      for (int nt = 0; nt < 4; ++nt) {
        int f = w * 64 + nt * 16 + m;
        const _Float16* bp = vT + (size_t)f * NN + jt + kk * 32 + koff;
        v16h b = load_v16h(bp, bp + 16);
        acc[nt] = __builtin_amdgcn_wmma_f32_16x16x32_f16(
            false, a, false, b, (short)0, acc[nt], false, false);
      }
    }
    __syncthreads();
  }

  // --- epilogue: normalize, skip + bias, ELU; emit f32 out + f16 copy
  float invl[8];
  #pragma unroll
  for (int r = 0; r < 8; ++r) invl[r] = 1.0f / sL[(lane < 16) ? r : r + 8];
  #pragma unroll
  for (int nt = 0; nt < 4; ++nt) {
    int f = w * 64 + nt * 16 + m;
    #pragma unroll
    for (int r = 0; r < 8; ++r) {
      int row = ibase + ((lane < 16) ? r : r + 8);
      float z = acc[nt][r] * invl[r] + X[(size_t)row * CC + f] + 1.0f;
      float u = z > 0.f ? z : __expf(z) - 1.0f;          // ELU
      outU[(size_t)row * CC + f] = u;
      U16[(size_t)row * CC + f] = (_Float16)u;
    }
  }
}

// ---------------------------------------------------------------- kernel 3
// sim = U U^T (WMMA), conn = sigmoid(sim)*(-dist)*deg (raw), + row mean/rstd.
__global__ __launch_bounds__(128) void k_conn(const _Float16* __restrict__ U16,
                                              const float* __restrict__ dist,
                                              const float* __restrict__ deg,
                                              float* __restrict__ conn,
                                              float* __restrict__ mu_out,
                                              float* __restrict__ rstd_out) {
  __shared__ float wsum[4][16], wsq[4][16];
  int t = threadIdx.x, lane = t & 31, w = t >> 5;
  int ibase = blockIdx.x * 16;
  int m = lane & 15;
  int aoff = (lane < 16) ? 0 : 8;
  int koff = (lane < 16) ? 0 : 8;

  // hoist all A fragments (16 rows x K=256) into registers for the j sweep
  v16h a[8];
  #pragma unroll
  for (int kk = 0; kk < 8; ++kk) {
    const _Float16* ap = U16 + (size_t)(ibase + m) * CC + kk * 32 + aoff;
    a[kk] = load_v16h(ap, ap + 16);
  }

  float rsum[8], rsq[8];
  #pragma unroll
  for (int r = 0; r < 8; ++r) { rsum[r] = 0.f; rsq[r] = 0.f; }

  for (int jb = 0; jb < NN; jb += 64) {
    int jcol = jb + w * 16;
    v8f acc = {};
    #pragma unroll
    for (int kk = 0; kk < 8; ++kk) {
      const _Float16* bp = U16 + (size_t)(jcol + m) * CC + kk * 32 + koff;
      v16h b = load_v16h(bp, bp + 16);
      acc = __builtin_amdgcn_wmma_f32_16x16x32_f16(
          false, a[kk], false, b, (short)0, acc, false, false);
    }
    #pragma unroll
    for (int r = 0; r < 8; ++r) {
      int row = ibase + ((lane < 16) ? r : r + 8);
      int j = jcol + m;
      size_t idx = (size_t)row * NN + j;
      float sg = 1.0f / (1.0f + __expf(-acc[r]));
      float cv = sg * (-dist[idx]) * deg[idx];
      conn[idx] = cv;
      rsum[r] += cv;
      rsq[r]  += cv * cv;
    }
  }

  // reduce across the 16 lanes that share each row
  #pragma unroll
  for (int r = 0; r < 8; ++r)
    #pragma unroll
    for (int msk = 8; msk >= 1; msk >>= 1) {
      rsum[r] += __shfl_xor(rsum[r], msk, 32);
      rsq[r]  += __shfl_xor(rsq[r], msk, 32);
    }
  if (m == 0) {
    int rbase = (lane < 16) ? 0 : 8;
    #pragma unroll
    for (int r = 0; r < 8; ++r) { wsum[w][rbase + r] = rsum[r]; wsq[w][rbase + r] = rsq[r]; }
  }
  __syncthreads();
  if (t < 16) {
    float s = 0.f, q = 0.f;
    #pragma unroll
    for (int ww = 0; ww < 4; ++ww) { s += wsum[ww][t]; q += wsq[ww][t]; }
    float mu  = s / (float)NN;
    float var = q / (float)NN - mu * mu;
    mu_out[ibase + t]   = mu;
    rstd_out[ibase + t] = rsqrtf(var + 1e-5f);
  }
}

// ---------------------------------------------------------------- kernel 4
// In-place layernorm + symmetrize: out[i,j]=out[j,i]=n(i,j)+n(j,i).
// Each (i<=j) pair owned by exactly one thread -> race-free in place.
__global__ __launch_bounds__(256) void k_sym(float* __restrict__ conn,
                                             const float* __restrict__ mu,
                                             const float* __restrict__ rstd) {
  int bi = blockIdx.x, bj = blockIdx.y;
  if (bi > bj) return;
  int t = threadIdx.x;
  int c = t & 31, r0 = t >> 5;
  #pragma unroll
  for (int k = 0; k < 4; ++k) {
    int i = bi * 32 + r0 + k * 8;
    int j = bj * 32 + c;
    if (i <= j) {
      size_t ij = (size_t)i * NN + j, ji = (size_t)j * NN + i;
      float na = (conn[ij] - mu[i]) * rstd[i];
      float nb = (conn[ji] - mu[j]) * rstd[j];
      float s = na + nb;
      conn[ij] = s;
      conn[ji] = s;
    }
  }
}

// ---------------------------------------------------------------- launch
extern "C" void kernel_launch(void* const* d_in, const int* in_sizes, int n_in,
                              void* d_out, int out_size, void* d_ws, size_t ws_size,
                              hipStream_t stream) {
  const float* X    = (const float*)d_in[0];
  const float* dist = (const float*)d_in[1];
  const float* bond = (const float*)d_in[2];
  /* d_in[3] edge_direction: provably unused (only l=0 SH path contributes) */
  const float* deg  = (const float*)d_in[4];
  const float* Wq   = (const float*)d_in[5];
  const float* Wk   = (const float*)d_in[6];
  const float* Wv   = (const float*)d_in[7];
  const float* wb   = (const float*)d_in[8];
  const float* wg   = (const float*)d_in[9];

  float* outU = (float*)d_out;                 // N*C updated nodes
  float* conn = outU + (size_t)NN * CC;        // N*N connectivity

  // workspace: [qk 16KB][mu 16KB][rstd 16KB][pad to 64KB][vT f16 2MB][U16 f16 2MB]
  float* ws_qk   = (float*)d_ws;
  float* ws_mu   = ws_qk + NN;
  float* ws_rstd = ws_mu + NN;
  _Float16* vT  = (_Float16*)((char*)d_ws + (64u << 10));
  _Float16* U16 = vT + (size_t)NN * CC;

  k_qk  <<<NN / 8, 256, 0, stream>>>(X, Wq, Wk, ws_qk);
  k_v   <<<NN / 16, 256, 0, stream>>>(X, Wv, vT);
  k_attn<<<NN / 16, 128, 0, stream>>>(X, dist, bond, deg, ws_qk, vT, wb, wg, outU, U16);
  k_conn<<<NN / 16, 128, 0, stream>>>(U16, dist, deg, conn, ws_mu, ws_rstd);
  dim3 g4(NN / 32, NN / 32);
  k_sym <<<g4, 256, 0, stream>>>(conn, ws_mu, ws_rstd);
}
